// EdgeReconstructionHead_29738353558366
// MI455X (gfx1250) — matrix operand order repaired and checked by reading, not compiled, not run on hardware
//
#include <hip/hip_runtime.h>
#include <hip/hip_bf16.h>

#define HID 128
#define K2H 256                 // 2*HID, GEMM K dimension
#define LDS_STRIDE 264          // padded column stride (bf16 elems) -> conflict-free B loads

typedef __attribute__((ext_vector_type(16))) __bf16 v16bf;
typedef __attribute__((ext_vector_type(8)))  __bf16 v8bf;
typedef __attribute__((ext_vector_type(8)))  float  v8f;
typedef __attribute__((ext_vector_type(4)))  float  f4v;

__device__ __forceinline__ v16bf load_a_frag(const float* __restrict__ er, int c0) {
    f4v f0 = *(const f4v*)(er + c0);
    f4v f1 = *(const f4v*)(er + c0 + 4);
    f4v f2 = *(const f4v*)(er + c0 + 8);
    f4v f3 = *(const f4v*)(er + c0 + 12);
    v16bf a;
#pragma unroll
    for (int j = 0; j < 4; ++j) {
        a[j]      = (__bf16)f0[j];
        a[4 + j]  = (__bf16)f1[j];
        a[8 + j]  = (__bf16)f2[j];
        a[12 + j] = (__bf16)f3[j];
    }
    return a;
}

__global__ __launch_bounds__(256)
void edge_head_wmma_kernel(const float* __restrict__ emb,        // [N_NODES, 128]
                           const long long* __restrict__ src_idx,// [E]
                           const long long* __restrict__ tgt_idx,// [E]
                           const float* __restrict__ W1,         // [256, 128] row-major
                           const float* __restrict__ b1,         // [128]
                           const float* __restrict__ W2,         // [128]
                           const float* __restrict__ b2,         // [1]
                           float* __restrict__ out,              // [E]
                           int n_edges)
{
    // W1 staged column-major bf16: w1s[n*LDS_STRIDE + k] = bf16(W1[k][n])
    __shared__ __align__(16) __bf16 w1s[HID * LDS_STRIDE];   // 67.6 KB of 320 KB WGP LDS

    const int tid  = threadIdx.x;
    const int lane = tid & 31;
    const int wave = tid >> 5;          // 0..7 waves per block
    const int ncol = lane & 15;         // N (or edge-row) index within 16
    const int khalf = (lane < 16) ? 0 : 1;

    // ---- One-time per block: stage W1 (f32 row-major) -> LDS (bf16 col-major) ----
    for (int idx = tid; idx < K2H * HID; idx += 256) {
        int k = idx >> 7;               // 0..255
        int n = idx & (HID - 1);        // 0..127
        w1s[n * LDS_STRIDE + k] = (__bf16)W1[idx];
    }
    __syncthreads();

    // ---- Per-lane constants (only ever indexed with unrolled/constant indices) ----
    float w2v[8], b1v[8];
#pragma unroll
    for (int nt = 0; nt < 8; ++nt) {
        w2v[nt] = W2[nt * 16 + ncol];
        b1v[nt] = b1[nt * 16 + ncol];
    }
    const float b2s = b2[0];

    // Each wave pass covers 32 edges: two 16-row A tiles sharing every B fragment read.
    const int groups = (n_edges + 31) >> 5;
#pragma unroll 1
    for (int g = blockIdx.x * 8 + wave; g < groups; g += gridDim.x * 8) {
        const int ebase = g << 5;
        int row0 = ebase + ncol;
        int row1 = ebase + 16 + ncol;
        int rc0 = (row0 < n_edges) ? row0 : (n_edges - 1);
        int rc1 = (row1 < n_edges) ? row1 : (n_edges - 1);

        const float* __restrict__ srow0 = emb + src_idx[rc0] * HID;
        const float* __restrict__ trow0 = emb + tgt_idx[rc0] * HID;
        const float* __restrict__ srow1 = emb + src_idx[rc1] * HID;
        const float* __restrict__ trow1 = emb + tgt_idx[rc1] * HID;

        // 2 x 8 persistent accumulator tiles, bias preloaded into C
        v8f acc0[8], acc1[8];
#pragma unroll
        for (int nt = 0; nt < 8; ++nt)
#pragma unroll
            for (int i = 0; i < 8; ++i) { acc0[nt][i] = b1v[nt]; acc1[nt][i] = b1v[nt]; }

        const int bbase = ncol * LDS_STRIDE + khalf * 16;

        // ---- K loop: two transient A fragments; each B fragment feeds two WMMAs ----
#pragma unroll 1
        for (int kc = 0; kc < 8; ++kc) {
            const int c0 = (kc & 3) * 32 + khalf * 16;
            v16bf a0 = load_a_frag((kc < 4) ? srow0 : trow0, c0);
            v16bf a1 = load_a_frag((kc < 4) ? srow1 : trow1, c0);

            const int koff = bbase + kc * 32;
#pragma unroll
            for (int nt = 0; nt < 8; ++nt) {
                // B fragment: 32 consecutive bytes in LDS (16B-aligned) -> 2x ds_load_b128
                const v8bf* bp = (const v8bf*)&w1s[koff + nt * (16 * LDS_STRIDE)];
                v8bf blo = bp[0];
                v8bf bhi = bp[1];
                v16bf b;
#pragma unroll
                for (int j = 0; j < 8; ++j) { b[j] = blo[j]; b[8 + j] = bhi[j]; }

                acc0[nt] = __builtin_amdgcn_wmma_f32_16x16x32_bf16(
                    false, a0, false, b, (short)0, acc0[nt], false, false);
                acc1[nt] = __builtin_amdgcn_wmma_f32_16x16x32_bf16(
                    false, a1, false, b, (short)0, acc1[nt], false, false);
            }
        }

        // ---- Fused epilogue: ReLU + dot with W2, then cross-lane reduce (per tile) ----
        float p0[8], p1[8];
#pragma unroll
        for (int i = 0; i < 8; ++i) { p0[i] = 0.f; p1[i] = 0.f; }
#pragma unroll
        for (int nt = 0; nt < 8; ++nt) {
#pragma unroll
            for (int i = 0; i < 8; ++i) {
                p0[i] += __builtin_fmaxf(acc0[nt][i], 0.f) * w2v[nt];
                p1[i] += __builtin_fmaxf(acc1[nt][i], 0.f) * w2v[nt];
            }
        }

        // Reduce over the 16 N-lanes of each half (C layout: lanes 0-15 rows 0..7, 16-31 rows 8..15)
#pragma unroll
        for (int i = 0; i < 8; ++i) {
            float v0 = p0[i], v1 = p1[i];
            v0 += __shfl_xor(v0, 1, 32);  v1 += __shfl_xor(v1, 1, 32);
            v0 += __shfl_xor(v0, 2, 32);  v1 += __shfl_xor(v1, 2, 32);
            v0 += __shfl_xor(v0, 4, 32);  v1 += __shfl_xor(v1, 4, 32);
            v0 += __shfl_xor(v0, 8, 32);  v1 += __shfl_xor(v1, 8, 32);
            p0[i] = v0 + b2s;
            p1[i] = v1 + b2s;
        }

        if ((lane & 15) == 0) {
            const int half8 = (lane < 16) ? 0 : 8;
            const int rb0 = ebase + half8;
            const int rb1 = ebase + 16 + half8;
            if (rb1 + 7 < n_edges) {
                f4v o;
                o = (f4v){p0[0], p0[1], p0[2], p0[3]}; *(f4v*)(out + rb0)     = o;
                o = (f4v){p0[4], p0[5], p0[6], p0[7]}; *(f4v*)(out + rb0 + 4) = o;
                o = (f4v){p1[0], p1[1], p1[2], p1[3]}; *(f4v*)(out + rb1)     = o;
                o = (f4v){p1[4], p1[5], p1[6], p1[7]}; *(f4v*)(out + rb1 + 4) = o;
            } else {
#pragma unroll
                for (int i = 0; i < 8; ++i) {
                    if (rb0 + i < n_edges) out[rb0 + i] = p0[i];
                    if (rb1 + i < n_edges) out[rb1 + i] = p1[i];
                }
            }
        }
    }
}

extern "C" void kernel_launch(void* const* d_in, const int* in_sizes, int n_in,
                              void* d_out, int out_size, void* d_ws, size_t ws_size,
                              hipStream_t stream) {
    const float*     emb = (const float*)d_in[0];
    const long long* si  = (const long long*)d_in[1];
    const long long* ti  = (const long long*)d_in[2];
    // d_in[3] = edge_type_idx (predictor already selected; unused)
    const float*     W1  = (const float*)d_in[4];
    const float*     b1  = (const float*)d_in[5];
    const float*     W2  = (const float*)d_in[6];
    const float*     b2  = (const float*)d_in[7];
    float* out = (float*)d_out;

    const int E = in_sizes[1];
    const int tiles = (E + 255) / 256;          // 256 edges per block pass
    int grid = tiles < 2048 ? tiles : 2048;     // persistent grid-stride
    if (grid < 1) grid = 1;

    edge_head_wmma_kernel<<<dim3(grid), dim3(256), 0, stream>>>(
        emb, si, ti, W1, b1, W2, b2, out, E);
}
